// VectorQuantizer_10703058502201
// MI455X (gfx1250) — matrix operand order repaired and checked by reading, not compiled, not run on hardware
//
#include <hip/hip_runtime.h>

// Problem constants (from reference): B=16, S=2048, D=512, K=8192
#define N_TOT   32768  // B*S rows
#define D_DIM   512
#define K_CODES 8192

typedef __attribute__((ext_vector_type(16))) __bf16 v16bf;
typedef __attribute__((ext_vector_type(8)))  __bf16 v8bf;
typedef __attribute__((ext_vector_type(8)))  float  v8f;

struct FragBF { v16bf hi; v16bf lo; };

__device__ __forceinline__ v8f bfwmma(v16bf a, v16bf b, v8f c) {
  // (neg_a, A, neg_b, B, c_mod, C, reuse_a, reuse_b)
  return __builtin_amdgcn_wmma_f32_16x16x32_bf16(false, a, false, b, (short)0, c,
                                                 false, false);
}

__device__ __forceinline__ unsigned orderKey(float f) {
  unsigned b = __float_as_uint(f);
  return (b & 0x80000000u) ? ~b : (b | 0x80000000u);  // monotone float->uint
}

// ---------------- Kernel: split f32 plane -> bf16 hi/lo planes ---------------
// Each thread handles 8 contiguous f32 -> one 16B hi store + one 16B lo store.
__global__ __launch_bounds__(256) void vq_split(const float* __restrict__ src,
                                                __bf16* __restrict__ hi,
                                                __bf16* __restrict__ lo) {
  int idx = blockIdx.x * blockDim.x + threadIdx.x;
  const float4* p = (const float4*)src + (size_t)idx * 2;
  float4 a = p[0], b = p[1];
  float t[8] = {a.x, a.y, a.z, a.w, b.x, b.y, b.z, b.w};
  v8bf h, l;
#pragma unroll
  for (int e = 0; e < 8; ++e) {
    float v = t[e];
    __bf16 hv = (__bf16)v;
    h[e] = hv;
    l[e] = (__bf16)(v - (float)hv);
  }
  *(v8bf*)(hi + (size_t)idx * 8) = h;
  *(v8bf*)(lo + (size_t)idx * 8) = l;
}

// ---------------- Kernel: per-code squared L2 norm (wave per code) -----------
__global__ __launch_bounds__(256) void vq_enorm(const float* __restrict__ emb,
                                                float* __restrict__ enorm) {
  int w = threadIdx.x >> 5, lane = threadIdx.x & 31;
  int code = blockIdx.x * 8 + w;
  const float4* p = (const float4*)(emb + (size_t)code * D_DIM + lane * 16);
  float s = 0.f;
#pragma unroll
  for (int i = 0; i < 4; ++i) {
    float4 v = p[i];
    s += v.x * v.x + v.y * v.y + v.z * v.z + v.w * v.w;
  }
#pragma unroll
  for (int m = 1; m <= 16; m <<= 1) s += __shfl_xor(s, m, 32);
  if (lane == 0) enorm[code] = s;
}

// ------------- Fragment loads from pre-split bf16 planes ---------------------
// A fragment 16x32 (MxK): lane half h holds k=16h+{0..7} (elems 0..7) and
// k=16h+8+{0..7} (elems 8..15) -> two contiguous 16B chunks of a bf16 row.
__device__ __forceinline__ v16bf loadA16(const __bf16* __restrict__ rowp, int half) {
  v8bf c0 = *(const v8bf*)(rowp + 16 * half);
  v8bf c1 = *(const v8bf*)(rowp + 16 * half + 8);
  return __builtin_shufflevector(c0, c1, 0, 1, 2, 3, 4, 5, 6, 7,
                                 8, 9, 10, 11, 12, 13, 14, 15);
}
// B fragment 32x16 (KxN): lane half h holds k=16h+{0..15} for column n=lane&15,
// i.e. one contiguous, 32B-aligned run of the bf16 embedding row.
__device__ __forceinline__ v16bf loadB16(const __bf16* __restrict__ p) {
  return *(const v16bf*)p;
}

// ------------- Main GEMM (pre-split planes) + partial argmin -----------------
// Grid: (N/128, K/128). Block: 256 threads = 8 waves in a 4(row) x 2(code)
// grid; each wave computes a 32x64 tile = 2x4 WMMA tiles over full D.
__global__ __launch_bounds__(256) void vq_gemm_pre(
    const __bf16* __restrict__ latHi, const __bf16* __restrict__ latLo,
    const __bf16* __restrict__ embHi, const __bf16* __restrict__ embLo,
    const float* __restrict__ enorm, unsigned long long* __restrict__ keys) {
  const int lane = threadIdx.x & 31;
  const int w = threadIdx.x >> 5;
  const int wr = w >> 1, wc = w & 1;
  const int half = lane >> 4, l16 = lane & 15;
  const int rowBase  = blockIdx.x * 128 + wr * 32;
  const int codeBase = blockIdx.y * 128 + wc * 64;

  v8f acc[2][4];
  const v8f z = {0.f, 0.f, 0.f, 0.f, 0.f, 0.f, 0.f, 0.f};
#pragma unroll
  for (int rt = 0; rt < 2; ++rt)
#pragma unroll
    for (int ct = 0; ct < 4; ++ct) acc[rt][ct] = z;

  const __bf16* a0h = latHi + (size_t)(rowBase + l16) * D_DIM;
  const __bf16* a0l = latLo + (size_t)(rowBase + l16) * D_DIM;
  const __bf16* a1h = latHi + (size_t)(rowBase + 16 + l16) * D_DIM;
  const __bf16* a1l = latLo + (size_t)(rowBase + 16 + l16) * D_DIM;

  for (int kb = 0; kb < D_DIM; kb += 32) {
    v16bf A0h = loadA16(a0h + kb, half);
    v16bf A0l = loadA16(a0l + kb, half);
    v16bf A1h = loadA16(a1h + kb, half);
    v16bf A1l = loadA16(a1l + kb, half);
#pragma unroll
    for (int ct = 0; ct < 4; ++ct) {
      size_t boff = (size_t)(codeBase + ct * 16 + l16) * D_DIM + kb + 16 * half;
      v16bf Bh = loadB16(embHi + boff);
      v16bf Bl = loadB16(embLo + boff);
      // x.e ~= xhi.ehi + xhi.elo + xlo.ehi   (~2^-16 effective precision)
      acc[0][ct] = bfwmma(A0h, Bh, acc[0][ct]);
      acc[0][ct] = bfwmma(A0h, Bl, acc[0][ct]);
      acc[0][ct] = bfwmma(A0l, Bh, acc[0][ct]);
      acc[1][ct] = bfwmma(A1h, Bh, acc[1][ct]);
      acc[1][ct] = bfwmma(A1h, Bl, acc[1][ct]);
      acc[1][ct] = bfwmma(A1l, Bh, acc[1][ct]);
    }
  }

  // Epilogue: minimize s = ||e||^2 - 2 x.e (||x||^2 is row-constant).
  float en[4];
#pragma unroll
  for (int ct = 0; ct < 4; ++ct) en[ct] = enorm[codeBase + ct * 16 + l16];

#pragma unroll
  for (int rt = 0; rt < 2; ++rt) {
#pragma unroll
    for (int j = 0; j < 8; ++j) {
      float best = en[0] - 2.0f * acc[rt][0][j];
      int bcode = codeBase + l16;
#pragma unroll
      for (int ct = 1; ct < 4; ++ct) {
        float s = en[ct] - 2.0f * acc[rt][ct][j];
        int c = codeBase + ct * 16 + l16;
        if (s < best || (s == best && c < bcode)) { best = s; bcode = c; }
      }
#pragma unroll
      for (int m = 1; m <= 8; m <<= 1) {  // xor masks stay within 16-lane half
        float ov = __shfl_xor(best, m, 32);
        int oc = __shfl_xor(bcode, m, 32);
        if (ov < best || (ov == best && oc < bcode)) { best = ov; bcode = oc; }
      }
      if (l16 == 0) {
        int row = rowBase + rt * 16 + 8 * half + j;
        unsigned long long packed =
            ((unsigned long long)orderKey(best) << 32) | (unsigned)bcode;
        atomicMin(&keys[row], packed);  // min value, then min index on ties
      }
    }
  }
}

// ------------- Fallback GEMM: on-the-fly split (small workspace) -------------
__device__ __forceinline__ void cvt_split16(const float* __restrict__ t, FragBF& f) {
#pragma unroll
  for (int e = 0; e < 16; ++e) {
    float v = t[e];
    __bf16 h = (__bf16)v;
    f.hi[e] = h;
    f.lo[e] = (__bf16)(v - (float)h);
  }
}
__device__ __forceinline__ FragBF loadA(const float* __restrict__ rowp, int half) {
  float t[16];
  const float4* p0 = (const float4*)(rowp + 16 * half);
  const float4* p1 = (const float4*)(rowp + 16 * half + 8);
  float4 a = p0[0], b = p0[1], c = p1[0], d = p1[1];
  t[0]=a.x; t[1]=a.y; t[2]=a.z; t[3]=a.w; t[4]=b.x; t[5]=b.y; t[6]=b.z; t[7]=b.w;
  t[8]=c.x; t[9]=c.y; t[10]=c.z; t[11]=c.w; t[12]=d.x; t[13]=d.y; t[14]=d.z; t[15]=d.w;
  FragBF f; cvt_split16(t, f); return f;
}
__device__ __forceinline__ FragBF loadB(const float* __restrict__ p) {
  float t[16];
  const float4* p4 = (const float4*)p;
  float4 a = p4[0], b = p4[1], c = p4[2], d = p4[3];
  t[0]=a.x; t[1]=a.y; t[2]=a.z; t[3]=a.w; t[4]=b.x; t[5]=b.y; t[6]=b.z; t[7]=b.w;
  t[8]=c.x; t[9]=c.y; t[10]=c.z; t[11]=c.w; t[12]=d.x; t[13]=d.y; t[14]=d.z; t[15]=d.w;
  FragBF f; cvt_split16(t, f); return f;
}

__global__ __launch_bounds__(256) void vq_gemm_fly(
    const float* __restrict__ lat, const float* __restrict__ emb,
    const float* __restrict__ enorm, unsigned long long* __restrict__ keys) {
  const int lane = threadIdx.x & 31;
  const int w = threadIdx.x >> 5;
  const int wr = w >> 1, wc = w & 1;
  const int half = lane >> 4, l16 = lane & 15;
  const int rowBase  = blockIdx.x * 128 + wr * 32;
  const int codeBase = blockIdx.y * 128 + wc * 64;

  v8f acc[2][4];
  const v8f z = {0.f, 0.f, 0.f, 0.f, 0.f, 0.f, 0.f, 0.f};
#pragma unroll
  for (int rt = 0; rt < 2; ++rt)
#pragma unroll
    for (int ct = 0; ct < 4; ++ct) acc[rt][ct] = z;

  const float* aRow0 = lat + (size_t)(rowBase + l16) * D_DIM;
  const float* aRow1 = lat + (size_t)(rowBase + 16 + l16) * D_DIM;

  for (int kb = 0; kb < D_DIM; kb += 32) {
    FragBF a0 = loadA(aRow0 + kb, half);
    FragBF a1 = loadA(aRow1 + kb, half);
#pragma unroll
    for (int ct = 0; ct < 4; ++ct) {
      const float* bp =
          emb + (size_t)(codeBase + ct * 16 + l16) * D_DIM + kb + 16 * half;
      FragBF b = loadB(bp);
      acc[0][ct] = bfwmma(a0.hi, b.hi, acc[0][ct]);
      acc[0][ct] = bfwmma(a0.hi, b.lo, acc[0][ct]);
      acc[0][ct] = bfwmma(a0.lo, b.hi, acc[0][ct]);
      acc[1][ct] = bfwmma(a1.hi, b.hi, acc[1][ct]);
      acc[1][ct] = bfwmma(a1.hi, b.lo, acc[1][ct]);
      acc[1][ct] = bfwmma(a1.lo, b.hi, acc[1][ct]);
    }
  }

  float en[4];
#pragma unroll
  for (int ct = 0; ct < 4; ++ct) en[ct] = enorm[codeBase + ct * 16 + l16];

#pragma unroll
  for (int rt = 0; rt < 2; ++rt) {
#pragma unroll
    for (int j = 0; j < 8; ++j) {
      float best = en[0] - 2.0f * acc[rt][0][j];
      int bcode = codeBase + l16;
#pragma unroll
      for (int ct = 1; ct < 4; ++ct) {
        float s = en[ct] - 2.0f * acc[rt][ct][j];
        int c = codeBase + ct * 16 + l16;
        if (s < best || (s == best && c < bcode)) { best = s; bcode = c; }
      }
#pragma unroll
      for (int m = 1; m <= 8; m <<= 1) {
        float ov = __shfl_xor(best, m, 32);
        int oc = __shfl_xor(bcode, m, 32);
        if (ov < best || (ov == best && oc < bcode)) { best = ov; bcode = oc; }
      }
      if (l16 == 0) {
        int row = rowBase + rt * 16 + 8 * half + j;
        unsigned long long packed =
            ((unsigned long long)orderKey(best) << 32) | (unsigned)bcode;
        atomicMin(&keys[row], packed);
      }
    }
  }
}

// ------------- Gather winner + straight-through f32 arithmetic ---------------
__global__ __launch_bounds__(128) void vq_gather(
    const float* __restrict__ lat, const float* __restrict__ emb,
    const unsigned long long* __restrict__ keys, float* __restrict__ out) {
  int n = blockIdx.x;
  int t = threadIdx.x;  // 128 threads * float4 = 512 floats
  unsigned code = (unsigned)(keys[n] & 0xFFFFFFFFull);
  const float4* q = (const float4*)(emb + (size_t)code * D_DIM);
  const float4* l = (const float4*)(lat + (size_t)n * D_DIM);
  float4* o = (float4*)(out + (size_t)n * D_DIM);
  float4 qv = q[t], lv = l[t], r;
  r.x = lv.x + (qv.x - lv.x);
  r.y = lv.y + (qv.y - lv.y);
  r.z = lv.z + (qv.z - lv.z);
  r.w = lv.w + (qv.w - lv.w);
  o[t] = r;
}

extern "C" void kernel_launch(void* const* d_in, const int* in_sizes, int n_in,
                              void* d_out, int out_size, void* d_ws, size_t ws_size,
                              hipStream_t stream) {
  const float* lat = (const float*)d_in[0];   // [16,2048,512] f32
  const float* emb = (const float*)d_in[1];   // [8192,512] f32
  float* out = (float*)d_out;                 // [16,2048,512] f32

  // Workspace layout
  const size_t offKeys  = 0;
  const size_t offEnorm = (size_t)N_TOT * 8;                       // 256 KB
  const size_t offLatHi = offEnorm + (size_t)K_CODES * 4;          // +32 KB
  const size_t latBytes = (size_t)N_TOT * D_DIM * 2;               // 32 MB
  const size_t embBytes = (size_t)K_CODES * D_DIM * 2;             // 8 MB
  const size_t offLatLo = offLatHi + latBytes;
  const size_t offEmbHi = offLatLo + latBytes;
  const size_t offEmbLo = offEmbHi + embBytes;
  const size_t needFast = offEmbLo + embBytes;                     // ~80.3 MB

  unsigned long long* keys = (unsigned long long*)((char*)d_ws + offKeys);
  float* enorm = (float*)((char*)d_ws + offEnorm);

  hipMemsetAsync(d_ws, 0xFF, (size_t)N_TOT * 8, stream);  // keys <- max u64
  vq_enorm<<<K_CODES / 8, 256, 0, stream>>>(emb, enorm);

  if (ws_size >= needFast) {
    __bf16* latHi = (__bf16*)((char*)d_ws + offLatHi);
    __bf16* latLo = (__bf16*)((char*)d_ws + offLatLo);
    __bf16* embHi = (__bf16*)((char*)d_ws + offEmbHi);
    __bf16* embLo = (__bf16*)((char*)d_ws + offEmbLo);
    // 8 elems/thread, 256 threads/block
    vq_split<<<(N_TOT * D_DIM) / (8 * 256), 256, 0, stream>>>(lat, latHi, latLo);
    vq_split<<<(K_CODES * D_DIM) / (8 * 256), 256, 0, stream>>>(emb, embHi, embLo);
    vq_gemm_pre<<<dim3(N_TOT / 128, K_CODES / 128), 256, 0, stream>>>(
        latHi, latLo, embHi, embLo, enorm, keys);
  } else {
    vq_gemm_fly<<<dim3(N_TOT / 128, K_CODES / 128), 256, 0, stream>>>(
        lat, emb, enorm, keys);
  }
  vq_gather<<<N_TOT, 128, 0, stream>>>(lat, emb, keys, out);
}
